// MultiHeadAttention2_76321568850505
// MI455X (gfx1250) — compile-verified, matrix operand order checked by reading, not run on hardware
//
#include <hip/hip_runtime.h>

#define B_  2
#define S_  2048
#define H_  1024
#define NH_ 16
#define P_  64

constexpr float SCALE = 0.125f;          // 1/sqrt(P)
constexpr float INV_S = 1.0f / (float)S_;

typedef __bf16 bf16;
typedef __attribute__((ext_vector_type(16))) __bf16 v16bf;
typedef __attribute__((ext_vector_type(8)))  float  v8f;

// float -> bf16, round-to-nearest-even
__device__ __forceinline__ bf16 f2bf(float f) {
  unsigned u = __float_as_uint(f);
  u += 0x7FFFu + ((u >> 16) & 1u);
  unsigned short h = (unsigned short)(u >> 16);
  return __builtin_bit_cast(bf16, h);
}

// A-fragment (16x32 MxK) / B-fragment (32x16 KxN) load:
// lane L reads 16 contiguous K-elems of row/col (L&15), K-offset (L>>4)*16.
__device__ __forceinline__ v16bf frag_ld(const bf16* base, int ld) {
  const int l = threadIdx.x & 31;
  return *(const v16bf*)(base + (l & 15) * ld + ((l >> 4) << 4));
}

__device__ __forceinline__ v8f wmma_bf16(v16bf a, v16bf b, v8f c) {
  return __builtin_amdgcn_wmma_f32_16x16x32_bf16(false, a, false, b, (short)0, c,
                                                 false, false);
}

// Async-copy one 64x32 bf16 weight tile (row stride in LDS = 48 elems / 96B)
// from global (row stride H_) into LDS. Block-cooperative: wave w moves rows
// 8w..8w+7; each lane moves one 16B chunk. Tracked by ASYNCcnt.
__device__ __forceinline__ void stage_async(bf16* tile, const bf16* Bbase, int h0,
                                            int wave, int l) {
  const int row   = wave * 8 + (l >> 2);    // 0..63
  const int chunk = l & 3;                  // 4 x 16B = 64B = 32 bf16 per row
  const bf16* gp = Bbase + (size_t)row * H_ + h0 + chunk * 8;
  unsigned ldsOff = (unsigned)(uintptr_t)(tile + row * 48 + chunk * 8);
  asm volatile("global_load_async_to_lds_b128 %0, %1, off"
               :: "v"(ldsOff), "v"((unsigned long long)(uintptr_t)gp)
               : "memory");
}
__device__ __forceinline__ void wait_async0() {
  asm volatile("s_wait_asynccnt 0x0" ::: "memory");
}

// ---------------- pack / convert kernels ----------------
__global__ void k_cvt(const float* __restrict__ in, bf16* __restrict__ out, int n) {
  int i = blockIdx.x * blockDim.x + threadIdx.x;
  if (i < n) out[i] = f2bf(in[i]);
}

// W (NH,3,H,P) f32  ->  Wt [(n*3+t)][p][h] bf16 (transposed: B-frags contiguous)
__global__ void k_packW(const float* __restrict__ W, bf16* __restrict__ Wt) {
  int i = blockIdx.x * blockDim.x + threadIdx.x;   // i = (nt*P + p)*H + h
  int h  = i & (H_ - 1);
  int p  = (i >> 10) & (P_ - 1);
  int nt = i >> 16;
  Wt[i] = f2bf(W[(size_t)(nt * H_ + h) * P_ + p]);
}

// ---------------- QKV projection ----------------
// grid.x = B*NH*3, grid.y = S/256, block=256. Wave: 32 s-rows x 64 p.
// Weight tiles staged block-wide in LDS via async-to-LDS, double-buffered;
// A fragments double-buffered in registers.
__global__ void k_proj(const bf16* __restrict__ xb, const bf16* __restrict__ Wt,
                       bf16* __restrict__ Kb, bf16* __restrict__ Qb,
                       bf16* __restrict__ Vt) {
  __shared__ __align__(32) bf16 tiles[2][64 * 48];
  const int idx = blockIdx.x;
  const int t   = idx % 3;
  const int n   = (idx / 3) % NH_;
  const int b   = idx / (3 * NH_);
  const int bn  = b * NH_ + n;
  const int wave = threadIdx.x >> 5;
  const int s0   = (blockIdx.y * 8 + wave) * 32;
  const int l    = threadIdx.x & 31;
  const int cn = l & 15, chi = (l >> 4) * 8;

  const bf16* Abase = xb + (size_t)(b * S_ + s0) * H_;
  const bf16* Bbase = Wt + (size_t)((n * 3 + t) * P_) * H_;

  stage_async(&tiles[0][0], Bbase, 0, wave, l);
  v16bf a0 = frag_ld(Abase, H_);
  v16bf a1 = frag_ld(Abase + (size_t)16 * H_, H_);
  wait_async0();
  __syncthreads();

  v8f acc0[4] = {}, acc1[4] = {};
  for (int h0 = 0, it = 0; h0 < H_; h0 += 32, ++it) {
    const int hn = (h0 + 32) & (H_ - 1);    // wraps on last iter (harmless)
    bf16* cur = &tiles[it & 1][0];
    bf16* nxt = &tiles[(it + 1) & 1][0];
    stage_async(nxt, Bbase, hn, wave, l);   // async fill of next buffer
    __builtin_prefetch(Abase + h0 + 256, 0, 1);
    v16bf a0n = frag_ld(Abase + hn, H_);
    v16bf a1n = frag_ld(Abase + (size_t)16 * H_ + hn, H_);
#pragma unroll
    for (int t4 = 0; t4 < 4; ++t4) {
      v16bf bbt = frag_ld(cur + t4 * 16 * 48, 48);
      acc0[t4] = wmma_bf16(a0, bbt, acc0[t4]);
      acc1[t4] = wmma_bf16(a1, bbt, acc1[t4]);
    }
    a0 = a0n; a1 = a1n;
    wait_async0();                           // own async writes landed
    __syncthreads();                         // publish next buffer block-wide
  }

  if (t == 1) {        // values -> store transposed Vt[bn][p][s]
#pragma unroll
    for (int t4 = 0; t4 < 4; ++t4)
#pragma unroll
      for (int r = 0; r < 8; ++r) {
        Vt[(size_t)(bn * P_ + t4 * 16 + cn) * S_ + s0 + chi + r]      = f2bf(acc0[t4][r]);
        Vt[(size_t)(bn * P_ + t4 * 16 + cn) * S_ + s0 + 16 + chi + r] = f2bf(acc1[t4][r]);
      }
  } else {             // keys / queries -> row-major [bn][s][p]
    bf16* dst = (t == 0) ? Kb : Qb;
#pragma unroll
    for (int t4 = 0; t4 < 4; ++t4)
#pragma unroll
      for (int r = 0; r < 8; ++r) {
        dst[(size_t)(bn * S_ + s0 + chi + r) * P_ + t4 * 16 + cn]      = f2bf(acc0[t4][r]);
        dst[(size_t)(bn * S_ + s0 + 16 + chi + r) * P_ + t4 * 16 + cn] = f2bf(acc1[t4][r]);
      }
  }
}

// ---------------- pass 1: per-column softmax stats over i ----------------
// grid.x = B*NH, grid.y = S/128; each wave owns 16 j-columns, loops all i.
__global__ void k_stats(const bf16* __restrict__ Kb, const bf16* __restrict__ Qb,
                        float* __restrict__ Mst, float* __restrict__ RL) {
  const int bn   = blockIdx.x;
  const int wave = threadIdx.x >> 5;
  const int j0   = (blockIdx.y * 8 + wave) * 16;
  const int l    = threadIdx.x & 31;

  const bf16* Qbase = Qb + (size_t)(bn * S_ + j0) * P_;
  v16bf bq0 = frag_ld(Qbase, P_);        // loop invariant
  v16bf bq1 = frag_ld(Qbase + 32, P_);

  const bf16* Kbase = Kb + (size_t)bn * S_ * P_;
  v16bf a0 = frag_ld(Kbase, P_);
  v16bf a1 = frag_ld(Kbase + 32, P_);

  float m = -3.0e38f, lsum = 0.0f;
  for (int i0 = 0; i0 < S_; i0 += 16) {
    const int in_ = (i0 + 16) & (S_ - 1);
    __builtin_prefetch(Kbase + (size_t)(i0 + 128) * P_, 0, 1);
    v16bf a0n = frag_ld(Kbase + (size_t)in_ * P_, P_);
    v16bf a1n = frag_ld(Kbase + (size_t)in_ * P_ + 32, P_);
    v8f c = {};
    c = wmma_bf16(a0, bq0, c);
    c = wmma_bf16(a1, bq1, c);
    float v[8];
#pragma unroll
    for (int r = 0; r < 8; ++r) v[r] = c[r] * SCALE;
    float tm = v[0];
#pragma unroll
    for (int r = 1; r < 8; ++r) tm = fmaxf(tm, v[r]);
    float mn = fmaxf(m, tm);
    float a = 0.0f;
#pragma unroll
    for (int r = 0; r < 8; ++r) a += __expf(v[r] - mn);
    lsum = lsum * __expf(m - mn) + a;
    m = mn;
    a0 = a0n; a1 = a1n;
  }
  float mo = __shfl_xor(m, 16, 32);
  float lo = __shfl_xor(lsum, 16, 32);
  float mn = fmaxf(m, mo);
  float lt = lsum * __expf(m - mn) + lo * __expf(mo - mn);
  if (l < 16) {
    Mst[bn * S_ + j0 + l] = mn;
    RL [bn * S_ + j0 + l] = 1.0f / lt;
  }
}

// ---------------- pass 2: out[i,p] = sum_j p_ij * V[j,p] ----------------
// grid.x = B*NH, grid.y = S/128; wave: 16 i-rows x 64 p, j in steps of 32.
// Q frags double-buffered; V frags issued early so exp/LDS overlap them.
__global__ void k_attn(const bf16* __restrict__ Kb, const bf16* __restrict__ Qb,
                       const bf16* __restrict__ Vt, const float* __restrict__ Mst,
                       const float* __restrict__ RL, const int* __restrict__ mask,
                       bf16* __restrict__ Hid) {
  __shared__ __align__(32) bf16 lds[8][16 * 48];   // per-wave 16x32 tile, stride 48
  const int bn   = blockIdx.x;
  const int b    = bn / NH_;
  const int n    = bn % NH_;
  const int wave = threadIdx.x >> 5;
  const int i0   = (blockIdx.y * 8 + wave) * 16;
  const int l    = threadIdx.x & 31;
  const int cn = l & 15, chi = (l >> 4) * 8;

  const bf16* Kbase = Kb + (size_t)(bn * S_ + i0) * P_;
  v16bf aK0 = frag_ld(Kbase, P_);        // loop invariant
  v16bf aK1 = frag_ld(Kbase + 32, P_);

  const bf16* Qall = Qb + (size_t)bn * S_ * P_;
  v16bf q[4];
  q[0] = frag_ld(Qall, P_);
  q[1] = frag_ld(Qall + 32, P_);
  q[2] = frag_ld(Qall + 16 * P_, P_);
  q[3] = frag_ld(Qall + 16 * P_ + 32, P_);

  bf16* myLds = &lds[wave][0];
  v8f o[4] = {};

  for (int j0 = 0; j0 < S_; j0 += 32) {
    // current-iteration V fragments (consumed after exp/LDS stage)
    v16bf bv[4];
#pragma unroll
    for (int t4 = 0; t4 < 4; ++t4)
      bv[t4] = frag_ld(Vt + (size_t)(bn * P_ + t4 * 16) * S_ + j0, S_);
    // next-iteration Q fragments (wrap-clamped)
    const int jn = (j0 + 32) & (S_ - 1);
    __builtin_prefetch(Vt + (size_t)(bn * P_) * S_ + j0 + 256, 0, 1);
    v16bf qn[4];
    qn[0] = frag_ld(Qall + (size_t)jn * P_, P_);
    qn[1] = frag_ld(Qall + (size_t)jn * P_ + 32, P_);
    qn[2] = frag_ld(Qall + (size_t)(jn + 16) * P_, P_);
    qn[3] = frag_ld(Qall + (size_t)(jn + 16) * P_ + 32, P_);

    v8f c0 = {}, c1 = {};
    c0 = wmma_bf16(aK0, q[0], c0);
    c0 = wmma_bf16(aK1, q[1], c0);
    c1 = wmma_bf16(aK0, q[2], c1);
    c1 = wmma_bf16(aK1, q[3], c1);

    const int col0 = j0 + cn, col1 = j0 + 16 + cn;
    float m0 = Mst[bn * S_ + col0], r0 = RL[bn * S_ + col0];
    float m1 = Mst[bn * S_ + col1], r1 = RL[bn * S_ + col1];
    int   k0 = mask[b * S_ + col0], k1 = mask[b * S_ + col1];
#pragma unroll
    for (int r = 0; r < 8; ++r) {
      // masked column j -> softmax over i of constant = 1/S
      float p0 = k0 ? __expf(c0[r] * SCALE - m0) * r0 : INV_S;
      float p1 = k1 ? __expf(c1[r] * SCALE - m1) * r1 : INV_S;
      myLds[(chi + r) * 48 + cn]      = f2bf(p0);
      myLds[(chi + r) * 48 + 16 + cn] = f2bf(p1);
    }
    __builtin_amdgcn_wave_barrier();     // wave-internal LDS RAW ordering
    v16bf aP = frag_ld(myLds, 48);       // C-layout -> A-layout via LDS
    __builtin_amdgcn_wave_barrier();
#pragma unroll
    for (int t4 = 0; t4 < 4; ++t4) o[t4] = wmma_bf16(aP, bv[t4], o[t4]);
#pragma unroll
    for (int t4 = 0; t4 < 4; ++t4) q[t4] = qn[t4];
  }
  // hidden[b][s][n*P + p]
#pragma unroll
  for (int t4 = 0; t4 < 4; ++t4)
#pragma unroll
    for (int r = 0; r < 8; ++r)
      Hid[(size_t)(b * S_ + i0 + chi + r) * H_ + n * P_ + t4 * 16 + cn] =
          f2bf(o[t4][r]);
}

// ---------------- final projection: out = hidden @ Wl^T + bl ----------------
// grid.x = (B*S)/256, grid.y = H/64. Wave: 32 rows x 64 o.
// Wl tiles staged in LDS via async-to-LDS, double-buffered.
__global__ void k_final(const bf16* __restrict__ Hid, const bf16* __restrict__ Wlb,
                        const float* __restrict__ bl, float* __restrict__ out) {
  __shared__ __align__(32) bf16 tiles[2][64 * 48];
  const int wave = threadIdx.x >> 5;
  const int row0 = (blockIdx.x * 8 + wave) * 32;   // row over B*S
  const int o0   = blockIdx.y * 64;
  const int l    = threadIdx.x & 31;
  const int cn = l & 15, chi = (l >> 4) * 8;

  const bf16* Abase = Hid + (size_t)row0 * H_;
  const bf16* Bbase = Wlb + (size_t)o0 * H_;

  stage_async(&tiles[0][0], Bbase, 0, wave, l);
  v16bf a0 = frag_ld(Abase, H_);
  v16bf a1 = frag_ld(Abase + (size_t)16 * H_, H_);
  wait_async0();
  __syncthreads();

  v8f acc0[4] = {}, acc1[4] = {};
  for (int h0 = 0, it = 0; h0 < H_; h0 += 32, ++it) {
    const int hn = (h0 + 32) & (H_ - 1);
    bf16* cur = &tiles[it & 1][0];
    bf16* nxt = &tiles[(it + 1) & 1][0];
    stage_async(nxt, Bbase, hn, wave, l);
    __builtin_prefetch(Abase + h0 + 256, 0, 1);
    v16bf a0n = frag_ld(Abase + hn, H_);
    v16bf a1n = frag_ld(Abase + (size_t)16 * H_ + hn, H_);
#pragma unroll
    for (int t4 = 0; t4 < 4; ++t4) {
      v16bf bbt = frag_ld(cur + t4 * 16 * 48, 48);
      acc0[t4] = wmma_bf16(a0, bbt, acc0[t4]);
      acc1[t4] = wmma_bf16(a1, bbt, acc1[t4]);
    }
    a0 = a0n; a1 = a1n;
    wait_async0();
    __syncthreads();
  }
#pragma unroll
  for (int t4 = 0; t4 < 4; ++t4) {
    float bias = bl[o0 + t4 * 16 + cn];
#pragma unroll
    for (int r = 0; r < 8; ++r) {
      out[(size_t)(row0 + chi + r) * H_ + o0 + t4 * 16 + cn]      = acc0[t4][r] + bias;
      out[(size_t)(row0 + 16 + chi + r) * H_ + o0 + t4 * 16 + cn] = acc1[t4][r] + bias;
    }
  }
}

// ---------------- launch ----------------
extern "C" void kernel_launch(void* const* d_in, const int* in_sizes, int n_in,
                              void* d_out, int out_size, void* d_ws, size_t ws_size,
                              hipStream_t stream) {
  const float* x    = (const float*)d_in[0];
  const int*   mask = (const int*)  d_in[1];
  const float* W    = (const float*)d_in[2];
  const float* Wl   = (const float*)d_in[3];
  const float* bl   = (const float*)d_in[4];
  float* out = (float*)d_out;

  char* w = (char*)d_ws;
  auto carve = [&](size_t bytes) {
    void* p = (void*)w;
    w += (bytes + 255) & ~(size_t)255;
    return p;
  };
  bf16*  xb  = (bf16*) carve((size_t)B_ * S_ * H_ * 2);
  bf16*  Wt  = (bf16*) carve((size_t)NH_ * 3 * H_ * P_ * 2);
  bf16*  Wlb = (bf16*) carve((size_t)H_ * H_ * 2);
  bf16*  Kb  = (bf16*) carve((size_t)B_ * NH_ * S_ * P_ * 2);
  bf16*  Qb  = (bf16*) carve((size_t)B_ * NH_ * S_ * P_ * 2);
  bf16*  Vt  = (bf16*) carve((size_t)B_ * NH_ * S_ * P_ * 2);
  float* Mst = (float*)carve((size_t)B_ * NH_ * S_ * 4);
  float* RL  = (float*)carve((size_t)B_ * NH_ * S_ * 4);
  bf16*  Hid = (bf16*) carve((size_t)B_ * S_ * H_ * 2);

  k_cvt  <<<(B_ * S_ * H_) / 256, 256, 0, stream>>>(x, xb, B_ * S_ * H_);
  k_cvt  <<<(H_ * H_) / 256, 256, 0, stream>>>(Wl, Wlb, H_ * H_);
  k_packW<<<(NH_ * 3 * H_ * P_) / 256, 256, 0, stream>>>(W, Wt);
  k_proj <<<dim3(B_ * NH_ * 3, S_ / 256), 256, 0, stream>>>(xb, Wt, Kb, Qb, Vt);
  k_stats<<<dim3(B_ * NH_, S_ / 128), 256, 0, stream>>>(Kb, Qb, Mst, RL);
  k_attn <<<dim3(B_ * NH_, S_ / 128), 256, 0, stream>>>(Kb, Qb, Vt, Mst, RL, mask, Hid);
  k_final<<<dim3((B_ * S_) / 256, H_ / 64), 256, 0, stream>>>(Hid, Wlb, bl, out);
}
